// RechitsGCN_8907762171916
// MI455X (gfx1250) — matrix operand order repaired and checked by reading, not compiled, not run on hardware
//
#include <hip/hip_runtime.h>

#define BB    16
#define NCL   32
#define NR    256
#define DIN   16
#define DOUT  64
#define NCLUST (BB*NCL)   // 512

typedef __attribute__((ext_vector_type(16))) __bf16 v16bf;
typedef __attribute__((ext_vector_type(8)))  float  v8f;

__device__ __forceinline__ __bf16 f2bf(float f){ return (__bf16)f; }
__device__ __forceinline__ float  bf2f(__bf16 b){ return (float)b; }

// A-fragment K offset for 16-bit WMMA: contiguous, element i of lane-group kh -> k = 16*kh + i
__device__ __forceinline__ int akoff(int i, int kh) {
    return 16 * kh + i;
}

// -------------------------------------------------------------------------
// Kernel 1: adj = exp(-dist) -> d_out, norm_k = (deg+1e-6)^-1.5 -> ws
// Thread t writes column t of every row (coalesced); symmetry gives in-degree.
// -------------------------------------------------------------------------
__global__ void k_adj(const float* __restrict__ x,
                      float* __restrict__ adj_out,
                      float* __restrict__ normk)
{
    const int cl  = blockIdx.x;
    const int tid = threadIdx.x;
    __shared__ float c0[NR], c1[NR];
    const float* xc = x + (size_t)cl * NR * DIN;
    c0[tid] = xc[tid * DIN + 0];
    c1[tid] = xc[tid * DIN + 1];
    __syncthreads();
    const float m0 = c0[tid], m1 = c1[tid];
    float* arow = adj_out + (size_t)cl * NR * NR;
    float deg = 0.0f;
    for (int r = 0; r < NR; ++r) {
        float dx = c0[r] - m0, dy = c1[r] - m1;
        float d2 = fminf(fmaxf(dx * dx + dy * dy, 1e-12f), 1e12f);
        float a = __expf(-sqrtf(d2));
        arow[r * NR + tid] = a;
        deg += a;
    }
    float nrm = rsqrtf(deg + 1e-6f);
    normk[cl * NR + tid] = nrm * nrm * nrm;   // norm^NCONV, NCONV=3
}

// -------------------------------------------------------------------------
// Kernel 2: f_het & gate per row -> LDS/ws (transposed),
//           f_hom = (adj^3 @ f_het*nk)*nk via WMMA, out_gcn -> ws,
//           q/k/v = out_gcn @ {Q,K,V} via WMMA (v stored transposed).
// -------------------------------------------------------------------------
__global__ void k_gcn_qkv(const float* __restrict__ x,
                          const float* __restrict__ W_t,
                          const float* __restrict__ b_t,
                          const float* __restrict__ theta,
                          const float* __restrict__ Q,
                          const float* __restrict__ K,
                          const float* __restrict__ V,
                          const float* __restrict__ adj,
                          const float* __restrict__ normk,
                          __bf16* __restrict__ gateT,   // [cl][n][m]
                          __bf16* __restrict__ og,      // [cl][m][n]
                          __bf16* __restrict__ qo,      // [cl][m][n]
                          __bf16* __restrict__ ko,      // [cl][m][n]
                          __bf16* __restrict__ vT)      // [cl][n][m]
{
    const int cl   = blockIdx.x;
    const int tid  = threadIdx.x;
    const int lane = tid & 31;
    const int wv   = tid >> 5;     // 8 waves
    const int kh   = lane >> 4;
    const int ln   = lane & 15;

    __shared__ __bf16 sh_fhsT[DOUT * NR];   // (f_het*nk) transposed: [n][m]
    __shared__ float  sh_nk[NR], sh_inv[NR];
    __shared__ __bf16 sh_th[DIN * DOUT], sh_wt[DIN * DOUT];
    __shared__ float  sh_bt[DOUT];

    for (int i = tid; i < DIN * DOUT; i += 256) {
        sh_th[i] = f2bf(theta[i]);
        sh_wt[i] = f2bf(W_t[i]);
    }
    if (tid < DOUT) sh_bt[tid] = b_t[tid];
    __syncthreads();

    // ---- phase 1: per-row f_het*nk (LDS, transposed) and gate (ws, transposed)
    const float* xr = x + (size_t)cl * NR * DIN + tid * DIN;
    float xv[DIN];
#pragma unroll
    for (int c = 0; c < DIN; ++c) xv[c] = xr[c];
    const float nk = normk[cl * NR + tid];
    sh_nk[tid]  = nk;
    sh_inv[tid] = 1.0f / nk;
    __bf16* gT = gateT + (size_t)cl * NR * DOUT;
    for (int d = 0; d < DOUT; ++d) {
        float s = 0.0f, g = sh_bt[d];
#pragma unroll
        for (int c = 0; c < DIN; ++c) {
            s += xv[c] * bf2f(sh_th[c * DOUT + d]);
            g += xv[c] * bf2f(sh_wt[c * DOUT + d]);
        }
        sh_fhsT[d * NR + tid] = f2bf(s * nk);
        gT[d * NR + tid] = f2bf(1.0f / (1.0f + __expf(-g)));
    }
    __syncthreads();

    // ---- phase 2: f_hom tiles + gate mix -> og (ws), one tile in flight
    const float* arow = adj + (size_t)cl * NR * NR;
    __bf16* ogc = og + (size_t)cl * NR * DOUT;
    for (int t = 0; t < 8; ++t) {
        const int mt = 2 * wv + (t >> 2);
        const int nt = t & 3;
        const int m  = mt * 16 + ln;
        const int n  = nt * 16 + ln;
        v8f acc = {};
#pragma unroll
        for (int kc = 0; kc < 8; ++kc) {
            v16bf a, b;
#pragma unroll
            for (int i = 0; i < 16; ++i) {
                const float av = arow[m * NR + kc * 32 + akoff(i, kh)];
                a[i] = f2bf(av * av * av);                       // adj^3
                b[i] = sh_fhsT[n * NR + kc * 32 + 16 * kh + i];  // contiguous
            }
            acc = __builtin_amdgcn_wmma_f32_16x16x32_bf16(false, a, false, b,
                                                          (short)0, acc, false, false);
        }
        const int mbase = mt * 16 + 8 * kh;
#pragma unroll
        for (int r = 0; r < 8; ++r) {
            const int mr = mbase + r;
            const float fhom = acc[r] * sh_nk[mr];
            const float fhet = bf2f(sh_fhsT[n * NR + mr]) * sh_inv[mr];
            const float g    = bf2f(gT[n * NR + mr]);
            ogc[mr * DOUT + n] = f2bf(g * fhom + (1.0f - g) * fhet);
        }
    }
    __threadfence_block();   // wave re-reads only its own rows of og

    // ---- phase 3: q/k/v
    const float* Ws[3] = { Q, K, V };
#pragma unroll
    for (int mtl = 0; mtl < 2; ++mtl) {
        const int mt = 2 * wv + mtl;
        const int m  = mt * 16 + ln;
        v16bf afr[2];
#pragma unroll
        for (int kc = 0; kc < 2; ++kc)
#pragma unroll
            for (int i = 0; i < 16; ++i)
                afr[kc][i] = ogc[m * DOUT + kc * 32 + akoff(i, kh)];
#pragma unroll
        for (int wsel = 0; wsel < 3; ++wsel) {
            const float* W = Ws[wsel];
#pragma unroll
            for (int nt = 0; nt < 4; ++nt) {
                const int n = nt * 16 + ln;
                v8f acc = {};
#pragma unroll
                for (int kc = 0; kc < 2; ++kc) {
                    v16bf b;
#pragma unroll
                    for (int i = 0; i < 16; ++i)
                        b[i] = f2bf(W[(kc * 32 + 16 * kh + i) * DOUT + n]);
                    acc = __builtin_amdgcn_wmma_f32_16x16x32_bf16(false, afr[kc], false, b,
                                                                  (short)0, acc, false, false);
                }
                const int mbase = mt * 16 + 8 * kh;
                if (wsel == 0) {
                    __bf16* dst = qo + (size_t)cl * NR * DOUT;
#pragma unroll
                    for (int r = 0; r < 8; ++r) dst[(mbase + r) * DOUT + n] = f2bf(acc[r]);
                } else if (wsel == 1) {
                    __bf16* dst = ko + (size_t)cl * NR * DOUT;
#pragma unroll
                    for (int r = 0; r < 8; ++r) dst[(mbase + r) * DOUT + n] = f2bf(acc[r]);
                } else {
                    __bf16* dst = vT + (size_t)cl * NR * DOUT;   // transposed
#pragma unroll
                    for (int r = 0; r < 8; ++r) dst[n * NR + mbase + r] = f2bf(acc[r]);
                }
            }
        }
    }
}

// -------------------------------------------------------------------------
// Kernel 3: logits = q @ k^T / 8, mask, softmax -> attn (d_out)
// Each wave owns complete 16-row strips => row stats via shfl in half-wave.
// -------------------------------------------------------------------------
__global__ void k_attn(const __bf16* __restrict__ q,
                       const __bf16* __restrict__ k,
                       const float* __restrict__ mask,
                       float* __restrict__ attn_out)
{
    const int cl   = blockIdx.x;
    const int tid  = threadIdx.x;
    const int lane = tid & 31;
    const int wv   = tid >> 5;
    const int kh   = lane >> 4;
    const int ln   = lane & 15;
    const __bf16* qc = q + (size_t)cl * NR * DOUT;
    const __bf16* kc = k + (size_t)cl * NR * DOUT;
    const float*  mk = mask + (size_t)cl * NR;
    float* ao = attn_out + (size_t)cl * NR * NR;

    for (int mtl = 0; mtl < 2; ++mtl) {
        const int mt = 2 * wv + mtl;
        v16bf afr[2];
#pragma unroll
        for (int c = 0; c < 2; ++c)
#pragma unroll
            for (int i = 0; i < 16; ++i)
                afr[c][i] = qc[(mt * 16 + ln) * DOUT + c * 32 + akoff(i, kh)];
        v8f   acc[16];
        float madd[16];
#pragma unroll
        for (int nt = 0; nt < 16; ++nt) {
            const int n = nt * 16 + ln;
            madd[nt] = (1.0f - mk[n]) * (-1e9f);
            v8f t = {};
#pragma unroll
            for (int c = 0; c < 2; ++c) {
                v16bf b;
#pragma unroll
                for (int i = 0; i < 16; ++i)
                    b[i] = kc[n * DOUT + c * 32 + 16 * kh + i];   // B[kk][n] = k[n][kk]
                t = __builtin_amdgcn_wmma_f32_16x16x32_bf16(false, afr[c], false, b,
                                                            (short)0, t, false, false);
            }
            acc[nt] = t;
        }
        float rmax[8], rsum[8];
#pragma unroll
        for (int r = 0; r < 8; ++r) { rmax[r] = -3.0e38f; rsum[r] = 0.0f; }
#pragma unroll
        for (int nt = 0; nt < 16; ++nt)
#pragma unroll
            for (int r = 0; r < 8; ++r) {
                const float l = acc[nt][r] * 0.125f + madd[nt];   // 1/sqrt(64)
                acc[nt][r] = l;
                rmax[r] = fmaxf(rmax[r], l);
            }
#pragma unroll
        for (int r = 0; r < 8; ++r)
#pragma unroll
            for (int off = 8; off >= 1; off >>= 1)
                rmax[r] = fmaxf(rmax[r], __shfl_xor(rmax[r], off, 32));
#pragma unroll
        for (int nt = 0; nt < 16; ++nt)
#pragma unroll
            for (int r = 0; r < 8; ++r) {
                const float e = __expf(acc[nt][r] - rmax[r]);
                acc[nt][r] = e;
                rsum[r] += e;
            }
#pragma unroll
        for (int r = 0; r < 8; ++r) {
#pragma unroll
            for (int off = 8; off >= 1; off >>= 1)
                rsum[r] += __shfl_xor(rsum[r], off, 32);
            rsum[r] = 1.0f / rsum[r];
        }
#pragma unroll
        for (int nt = 0; nt < 16; ++nt) {
            const int n = nt * 16 + ln;
#pragma unroll
            for (int r = 0; r < 8; ++r)
                ao[(mt * 16 + r + 8 * kh) * NR + n] = acc[nt][r] * rsum[r];
        }
    }
}

// -------------------------------------------------------------------------
// Kernel 4: sa = attn @ v -> d_out (also staging),
//           convout = relu(sa @ W_out + b_out), output = sum_rows(convout*mask)
// -------------------------------------------------------------------------
__global__ void k_out(const __bf16* __restrict__ vT,
                      const float* __restrict__ W_out,
                      const float* __restrict__ b_out,
                      const float* __restrict__ mask,
                      const float* __restrict__ attn,
                      float* __restrict__ sa_out,
                      float* __restrict__ out)
{
    const int cl   = blockIdx.x;
    const int tid  = threadIdx.x;
    const int lane = tid & 31;
    const int wv   = tid >> 5;
    const int kh   = lane >> 4;
    const int ln   = lane & 15;
    __shared__ float cs[DOUT];
    if (tid < DOUT) cs[tid] = 0.0f;
    __syncthreads();

    const float*  ac = attn + (size_t)cl * NR * NR;
    const __bf16* vc = vT + (size_t)cl * NR * DOUT;   // [n][k]
    const float*  mk = mask + (size_t)cl * NR;
    float* sac = sa_out + (size_t)cl * NR * DOUT;

    for (int mtl = 0; mtl < 2; ++mtl) {
        const int mt = 2 * wv + mtl;
        const int m  = mt * 16 + ln;
#pragma unroll
        for (int nt = 0; nt < 4; ++nt) {
            const int n = nt * 16 + ln;
            v8f acc = {};
#pragma unroll
            for (int kcc = 0; kcc < 8; ++kcc) {
                v16bf a, b;
#pragma unroll
                for (int i = 0; i < 16; ++i) {
                    a[i] = f2bf(ac[m * NR + kcc * 32 + akoff(i, kh)]);
                    b[i] = vc[n * NR + kcc * 32 + 16 * kh + i];   // contiguous
                }
                acc = __builtin_amdgcn_wmma_f32_16x16x32_bf16(false, a, false, b,
                                                              (short)0, acc, false, false);
            }
#pragma unroll
            for (int r = 0; r < 8; ++r)
                sac[(mt * 16 + r + 8 * kh) * DOUT + n] = acc[r];
        }
        __threadfence_block();   // strip of sa visible across lanes of this wave
#pragma unroll
        for (int nt = 0; nt < 4; ++nt) {
            const int n = nt * 16 + ln;
            v8f acc = {};
#pragma unroll
            for (int kcc = 0; kcc < 2; ++kcc) {
                v16bf a, b;
#pragma unroll
                for (int i = 0; i < 16; ++i) {
                    a[i] = f2bf(sac[m * DOUT + kcc * 32 + akoff(i, kh)]);
                    b[i] = f2bf(W_out[(kcc * 32 + 16 * kh + i) * DOUT + n]);
                }
                acc = __builtin_amdgcn_wmma_f32_16x16x32_bf16(false, a, false, b,
                                                              (short)0, acc, false, false);
            }
            const float bo = b_out[n];
            float ps = 0.0f;
#pragma unroll
            for (int r = 0; r < 8; ++r) {
                const int mr = mt * 16 + r + 8 * kh;
                ps += fmaxf(acc[r] + bo, 0.0f) * mk[mr];
            }
            ps += __shfl_xor(ps, 16, 32);        // fold both half-wave row groups
            if (kh == 0) atomicAdd(&cs[n], ps);  // ds_add_f32
        }
    }
    __syncthreads();
    if (tid < DOUT) out[(size_t)cl * DOUT + tid] = cs[tid];
}

// -------------------------------------------------------------------------
extern "C" void kernel_launch(void* const* d_in, const int* in_sizes, int n_in,
                              void* d_out, int out_size, void* d_ws, size_t ws_size,
                              hipStream_t stream)
{
    (void)in_sizes; (void)n_in; (void)out_size; (void)ws_size;
    const float* x     = (const float*)d_in[0];
    const float* mask  = (const float*)d_in[1];
    const float* W_t   = (const float*)d_in[2];
    const float* b_t   = (const float*)d_in[3];
    const float* theta = (const float*)d_in[4];
    const float* Q     = (const float*)d_in[5];
    const float* K     = (const float*)d_in[6];
    const float* V     = (const float*)d_in[7];
    const float* W_out = (const float*)d_in[8];
    const float* b_out = (const float*)d_in[9];

    // d_out layout (return order): output, sa_output, attn, adj
    float* out  = (float*)d_out;
    float* sa   = out  + (size_t)NCLUST * DOUT;
    float* attn = sa   + (size_t)NCLUST * NR * DOUT;
    float* adj  = attn + (size_t)NCLUST * NR * NR;

    // ws layout: normk f32 | gateT | og | q | k | vT (bf16, NCLUST*NR*DOUT each)
    const size_t MATSZ = (size_t)NCLUST * NR * DOUT;
    float*  normk = (float*)d_ws;
    __bf16* gateT = (__bf16*)(normk + (size_t)NCLUST * NR);
    __bf16* og    = gateT + MATSZ;
    __bf16* qb    = og + MATSZ;
    __bf16* kb    = qb + MATSZ;
    __bf16* vTb   = kb + MATSZ;

    dim3 grid(NCLUST), block(256);
    k_adj    <<<grid, block, 0, stream>>>(x, adj, normk);
    k_gcn_qkv<<<grid, block, 0, stream>>>(x, W_t, b_t, theta, Q, K, V, adj, normk,
                                          gateT, og, qb, kb, vTb);
    k_attn   <<<grid, block, 0, stream>>>(qb, kb, mask, attn);
    k_out    <<<grid, block, 0, stream>>>(vTb, W_out, b_out, mask, attn, sa, out);
}
